// ComparisonLayer_11854109737245
// MI455X (gfx1250) — matrix operand order repaired and checked by reading, not compiled, not run on hardware
//
#include <hip/hip_runtime.h>
#include <hip/hip_bf16.h>

typedef __attribute__((ext_vector_type(16))) __bf16 v16bf;
typedef __attribute__((ext_vector_type(8)))  __bf16 v8bf;
typedef __attribute__((ext_vector_type(8)))  float  v8f;

#define LL 512
#define DD 256
#define CC 128
#define OO 64

// ---------------------------------------------------------------------------
// Kernel 1: layernorm + projections a,b (bf16) + factored diff-term GEMVs.
// One block per (b,l) row; 256 threads.
// ---------------------------------------------------------------------------
__global__ void cl_prep(const float* __restrict__ x,  const float* __restrict__ nw,
                        const float* __restrict__ nb, const float* __restrict__ w1,
                        const float* __restrict__ b1, const float* __restrict__ w2,
                        const float* __restrict__ b2, const float* __restrict__ b3,
                        const float* __restrict__ w4,
                        __bf16* __restrict__ a_bf, __bf16* __restrict__ b_bf,
                        float* __restrict__ A4, float* __restrict__ B4) {
    __shared__ float s1[DD];
    __shared__ float xn[DD];
    __shared__ float av[CC];
    __shared__ float bv[CC];
    const int t   = threadIdx.x;
    const int row = blockIdx.x;            // row = b*512 + l
    const int bb  = row >> 9;
    const int l   = row & (LL - 1);

    // x is [L, B, D]; reference transposes to [B, L, D]
    const float v = x[(l * 2 + bb) * DD + t];

    s1[t] = v; __syncthreads();
    #pragma unroll
    for (int s = 128; s > 0; s >>= 1) { if (t < s) s1[t] += s1[t + s]; __syncthreads(); }
    const float mu = s1[0] * (1.0f / 256.0f);
    __syncthreads();
    const float d = v - mu;
    s1[t] = d * d; __syncthreads();
    #pragma unroll
    for (int s = 128; s > 0; s >>= 1) { if (t < s) s1[t] += s1[t + s]; __syncthreads(); }
    const float var = s1[0] * (1.0f / 256.0f);

    xn[t] = d * rsqrtf(var + 1e-5f) * nw[t] + nb[t];
    __syncthreads();

    // projections: thread t<128 -> a[c=t], t>=128 -> b[c=t-128]
    if (t < CC) {
        const float* wr = w1 + t * DD;
        float acc = b1[t];
        #pragma unroll 8
        for (int dd = 0; dd < DD; ++dd) acc += xn[dd] * wr[dd];
        av[t] = acc;
        a_bf[(size_t)row * CC + t] = (__bf16)acc;
    } else {
        const int c = t - CC;
        const float* wr = w2 + c * DD;
        float acc = b2[c];
        #pragma unroll 8
        for (int dd = 0; dd < DD; ++dd) acc += xn[dd] * wr[dd];
        bv[c] = acc;
        b_bf[(size_t)row * CC + c] = (__bf16)acc;
    }
    __syncthreads();

    // diff-term factorization:  A4 = a@w4^T + b3 ,  B4 = b@w4^T
    if (t < OO) {
        const float* wr = w4 + t * CC;
        float acc = b3[t];
        #pragma unroll 8
        for (int c = 0; c < CC; ++c) acc += av[c] * wr[c];
        A4[(size_t)row * OO + t] = acc;
    } else if (t < 2 * OO) {
        const int o = t - OO;
        const float* wr = w4 + o * CC;
        float acc = 0.0f;
        #pragma unroll 8
        for (int c = 0; c < CC; ++c) acc += bv[c] * wr[c];
        B4[(size_t)row * OO + o] = acc;
    }
}

// ---------------------------------------------------------------------------
// Kernel 2: per-(b,i) GEMM  Out_i[j,o] = B_mat[j,c] x (w3 .* a_i)[c,o]
// via v_wmma_f32_16x16x32_bf16, epilogue adds A4[i,o] - B4[j,o].
// grid.x = b*512+i (1024), grid.y = j-block of 128; 8 waves, each one
// 16(j) x 64(o) tile.
// ---------------------------------------------------------------------------
__device__ __forceinline__ v8f wmma_step(const __bf16* p, v16bf af, v8f acc) {
    const v8bf* q = (const v8bf*)p;            // 32 contiguous bytes in LDS
    v8bf l0 = q[0], l1 = q[1];
    v16bf bf_ = __builtin_shufflevector(l0, l1, 0,1,2,3,4,5,6,7,8,9,10,11,12,13,14,15);
    return __builtin_amdgcn_wmma_f32_16x16x32_bf16(false, af, false, bf_,
                                                   (short)0, acc, false, false);
}

__global__ void cl_main(const __bf16* __restrict__ b_bf, const __bf16* __restrict__ a_bf,
                        const float* __restrict__ w3,   const float* __restrict__ A4,
                        const float* __restrict__ B4,   float* __restrict__ out) {
    __shared__ __align__(16) __bf16 w3a[OO * CC];   // 16 KB: (w3 .* a_i), [o][c]
    const int tid = threadIdx.x;
    const int row = blockIdx.x;                     // b*512 + i
    const int bb  = row >> 9;

    // Build scaled weight tile once per workgroup, reused by all 8 waves.
    const __bf16* arow = a_bf + (size_t)row * CC;
    for (int idx = tid; idx < OO * CC; idx += 256) {
        const int c = idx & (CC - 1);
        w3a[idx] = (__bf16)(w3[idx] * (float)arow[c]);
    }
    __syncthreads();

    const int wave  = tid >> 5;
    const int lane  = tid & 31;
    const int lg    = lane >> 4;                    // lane group (0/1)
    const int nn    = lane & 15;
    const int jbase = blockIdx.y * 128 + wave * 16;

    v8f acc0 = {}, acc1 = {}, acc2 = {}, acc3 = {};

    // A-matrix (M=j rows of b_bf): lane holds row M=nn; K chunks (contiguous,
    // matching B): group0 -> K {0..15}, group1 -> K {16..31}.
    const __bf16* brow = b_bf + ((size_t)(bb * LL + jbase + nn)) * CC;
    const int k0 = lg * 16;

    #pragma unroll
    for (int cb = 0; cb < CC; cb += 32) {
        v8bf lo = *(const v8bf*)(brow + cb + k0);
        v8bf hi = *(const v8bf*)(brow + cb + k0 + 8);
        v16bf af = __builtin_shufflevector(lo, hi, 0,1,2,3,4,5,6,7,8,9,10,11,12,13,14,15);
        const int kb = cb + lg * 16;                // B layout: K=(lane>>4)*16+e
        acc0 = wmma_step(&w3a[( 0 + nn) * CC + kb], af, acc0);
        acc1 = wmma_step(&w3a[(16 + nn) * CC + kb], af, acc1);
        acc2 = wmma_step(&w3a[(32 + nn) * CC + kb], af, acc2);
        acc3 = wmma_step(&w3a[(48 + nn) * CC + kb], af, acc3);
    }

    // Epilogue: C/D layout -> element r is (M = r + 8*lg, N = nn).
    const int jr = jbase + lg * 8;
#define STORE_TILE(T, ACC)                                                     \
    {                                                                          \
        const int ob = (T)*16 + nn;                                            \
        const float a4v = A4[(size_t)row * OO + ob];                           \
        const float* b4p = B4 + ((size_t)(bb * LL + jr)) * OO + ob;            \
        float* op = out + ((size_t)row * LL + jr) * OO + ob;                   \
        _Pragma("unroll")                                                      \
        for (int r = 0; r < 8; ++r)                                            \
            op[(size_t)r * OO] = ACC[r] + a4v - b4p[(size_t)r * OO];           \
    }
    STORE_TILE(0, acc0)
    STORE_TILE(1, acc1)
    STORE_TILE(2, acc2)
    STORE_TILE(3, acc3)
#undef STORE_TILE
}

// ---------------------------------------------------------------------------
extern "C" void kernel_launch(void* const* d_in, const int* in_sizes, int n_in,
                              void* d_out, int out_size, void* d_ws, size_t ws_size,
                              hipStream_t stream) {
    const float* x      = (const float*)d_in[0];
    const float* norm_w = (const float*)d_in[1];
    const float* norm_b = (const float*)d_in[2];
    const float* w1     = (const float*)d_in[3];
    const float* b1     = (const float*)d_in[4];
    const float* w2     = (const float*)d_in[5];
    const float* b2     = (const float*)d_in[6];
    const float* w3     = (const float*)d_in[7];
    const float* b3     = (const float*)d_in[8];
    const float* w4     = (const float*)d_in[9];
    float* out = (float*)d_out;

    char* ws = (char*)d_ws;
    __bf16* a_bf = (__bf16*)(ws);                    // 1024*128*2 = 256 KB
    __bf16* b_bf = (__bf16*)(ws + 256 * 1024);       // 256 KB
    float*  A4   = (float*)(ws + 512 * 1024);        // 1024*64*4 = 256 KB
    float*  B4   = (float*)(ws + 768 * 1024);        // 256 KB

    cl_prep<<<dim3(2 * LL), dim3(256), 0, stream>>>(x, norm_w, norm_b, w1, b1,
                                                    w2, b2, b3, w4,
                                                    a_bf, b_bf, A4, B4);

    cl_main<<<dim3(2 * LL, LL / 128), dim3(256), 0, stream>>>(b_bf, a_bf, w3,
                                                              A4, B4, out);
}
